// SpatialBiasedAttention_90486370992268
// MI455X (gfx1250) — matrix operand order repaired and checked by reading, not compiled, not run on hardware
//
#include <hip/hip_runtime.h>
#include <hip/hip_bf16.h>

// ---------------- CDNA5 WMMA types ----------------
typedef __bf16 v16bf __attribute__((ext_vector_type(16)));
typedef float  v8f   __attribute__((ext_vector_type(8)));

__device__ __forceinline__ v8f vzero() {
  v8f z;
#pragma unroll
  for (int i = 0; i < 8; ++i) z[i] = 0.f;
  return z;
}

__device__ __forceinline__ v8f wmma_bf16(v16bf a, v16bf b, v8f c) {
  // D = A(16x32 bf16) x B(32x16 bf16) + C(16x16 f32)
  return __builtin_amdgcn_wmma_f32_16x16x32_bf16(
      /*neg_a=*/false, a, /*neg_b=*/false, b,
      /*c_mod=*/(short)0, c, /*reuse_a=*/false, /*reuse_b=*/false);
}

// ---- CDNA5 async global->LDS (ASYNCcnt path, CDNA5 ISA) ----
// Per-lane: copies 16 bytes from gaddr to LDS offset lds_off, tracked by ASYNCcnt.
__device__ __forceinline__ void async_load_b128(unsigned lds_off, const void* gaddr) {
  asm volatile("global_load_async_to_lds_b128 %0, %1, off"
               :: "v"(lds_off), "v"(gaddr) : "memory");
}
__device__ __forceinline__ void wait_async0() {
  asm volatile("s_wait_asynccnt 0" ::: "memory");
}

// A fragment: 16x32 (MxK), 16-bit. Lane layout used here (contiguous k = 16*half + s):
//  lanes 0-15 / 16-31 select K halves; VGPR v holds K pairs.
__device__ __forceinline__ v16bf frag_a(const __bf16* base, int stride, int row, int hb) {
  v16bf a;
#pragma unroll
  for (int v = 0; v < 8; ++v) {
    int kk = (v << 1) + (hb << 4);
    a[2 * v]     = base[row * stride + kk];
    a[2 * v + 1] = base[row * stride + kk + 1];
  }
  return a;
}

// B fragment from transposed storage base[col][k] (K^T / weight tiles).
__device__ __forceinline__ v16bf frag_bt(const __bf16* base, int stride, int col, int hb) {
  v16bf b;
#pragma unroll
  for (int v = 0; v < 8; ++v) {
    int kk = (v << 1) + (hb << 4);
    b[2 * v]     = base[col * stride + kk];
    b[2 * v + 1] = base[col * stride + kk + 1];
  }
  return b;
}

// B fragment from natural storage base[k][col] (V tiles).
__device__ __forceinline__ v16bf frag_bn(const __bf16* base, int stride, int col, int hb) {
  v16bf b;
#pragma unroll
  for (int v = 0; v < 8; ++v) {
    int kk = (v << 1) + (hb << 4);
    b[2 * v]     = base[kk * stride + col];
    b[2 * v + 1] = base[(kk + 1) * stride + col];
  }
  return b;
}

// ---------------- problem constants ----------------
#define BB 2
#define NN 2048
#define DD 1024
#define HH 16
#define HD 64
#define PP 1024
#define SCALE 0.125f
#define BIAS_STRENGTH 0.1f

// ---------------- small prep kernels ----------------
__global__ void init_kernel(unsigned* maxd) { *maxd = 0u; }

__global__ void __launch_bounds__(256) f32_to_bf16_kernel(const float* __restrict__ s,
                                                          __bf16* __restrict__ d, int n) {
  int i = blockIdx.x * 256 + threadIdx.x;
  if (i < n) d[i] = (__bf16)s[i];
}

__global__ void __launch_bounds__(256) rowstats_kernel(const float* __restrict__ x,
                                                       float* __restrict__ stats) {
  __shared__ float sh0[256], sh1[256];
  int row = blockIdx.x;
  const float* xr = x + (size_t)row * DD;
  float s = 0.f, s2 = 0.f;
  for (int i = threadIdx.x; i < DD; i += 256) { float v = xr[i]; s += v; s2 += v * v; }
  sh0[threadIdx.x] = s; sh1[threadIdx.x] = s2;
  __syncthreads();
  for (int off = 128; off > 0; off >>= 1) {
    if ((int)threadIdx.x < off) {
      sh0[threadIdx.x] += sh0[threadIdx.x + off];
      sh1[threadIdx.x] += sh1[threadIdx.x + off];
    }
    __syncthreads();
  }
  if (threadIdx.x == 0) {
    float mu  = sh0[0] * (1.f / DD);
    float var = sh1[0] * (1.f / DD) - mu * mu;
    stats[2 * row]     = mu;
    stats[2 * row + 1] = rsqrtf(var + 1e-5f);
  }
}

__global__ void __launch_bounds__(256) maxdist_kernel(const float* __restrict__ c,
                                                      unsigned* __restrict__ maxd) {
  int idx = blockIdx.x * 256 + threadIdx.x;  // P*P
  int i = idx >> 10, j = idx & (PP - 1);
  float dx = c[2 * i] - c[2 * j];
  float dy = c[2 * i + 1] - c[2 * j + 1];
  float d = sqrtf(dx * dx + dy * dy);
#pragma unroll
  for (int off = 16; off >= 1; off >>= 1) d = fmaxf(d, __shfl_xor(d, off, 32));
  if ((threadIdx.x & 31) == 0) atomicMax(maxd, __float_as_uint(d));  // d >= 0
}

__global__ void __launch_bounds__(256) bias_kernel(const float* __restrict__ c,
                                                   const unsigned* __restrict__ maxd,
                                                   float* __restrict__ bias) {
  size_t idx = (size_t)blockIdx.x * 256 + threadIdx.x;  // N*N
  int i = (int)(idx >> 11);
  int j = (int)(idx & (NN - 1));
  float md = __uint_as_float(*maxd);
  float v = 0.f;
  if (i >= 1 && i <= PP && j >= 1 && j <= PP && md > 1e-6f) {
    float dx = c[2 * (i - 1)] - c[2 * (j - 1)];
    float dy = c[2 * (i - 1) + 1] - c[2 * (j - 1) + 1];
    v = -sqrtf(dx * dx + dy * dy) / md * BIAS_STRENGTH;
  }
  bias[idx] = v;
}

// ---------------- QKV GEMM: qkv = x @ Wqkv^T, scatter into per-head Q/K/V ----
// M=4096, Nout=3072, K=1024. 128x128 tile, 8 waves (2x4), wave = 64x32.
// Double-buffered LDS, tiles staged with global_load_async_to_lds_b128.
__global__ void __launch_bounds__(256) qkv_gemm_kernel(
    const __bf16* __restrict__ A, const __bf16* __restrict__ Bw,
    __bf16* __restrict__ Qb, __bf16* __restrict__ Kb, __bf16* __restrict__ Vb) {
  const int tid = threadIdx.x, lane = tid & 31, wave = tid >> 5;
  const int ll = lane & 15, hb = lane >> 4;
  const int wm = wave >> 2, wn = wave & 3;
  const int m0 = blockIdx.x * 128, n0 = blockIdx.y * 128;

  __shared__ __bf16 lA[2][128 * 40];
  __shared__ __bf16 lB[2][128 * 40];

  auto stage = [&](int kt, int buf) {
    for (int vv = tid; vv < 512; vv += 256) {
      int row = vv >> 2, kv = (vv & 3) << 3;
      async_load_b128((unsigned)(size_t)&lA[buf][row * 40 + kv],
                      A + (size_t)(m0 + row) * DD + kt + kv);
      async_load_b128((unsigned)(size_t)&lB[buf][row * 40 + kv],
                      Bw + (size_t)(n0 + row) * DD + kt + kv);
    }
  };

  v8f acc[4][2];
#pragma unroll
  for (int mf = 0; mf < 4; ++mf)
#pragma unroll
    for (int nf = 0; nf < 2; ++nf) acc[mf][nf] = vzero();

  stage(0, 0);
  wait_async0();
  __syncthreads();

  for (int t = 0; t < DD / 32; ++t) {
    if (t + 1 < DD / 32) stage((t + 1) * 32, (t + 1) & 1);
    const __bf16* cA = lA[t & 1];
    const __bf16* cB = lB[t & 1];

    v16bf af[4];
#pragma unroll
    for (int mf = 0; mf < 4; ++mf) af[mf] = frag_a(cA, 40, wm * 64 + mf * 16 + ll, hb);
    v16bf bfr[2];
#pragma unroll
    for (int nf = 0; nf < 2; ++nf) bfr[nf] = frag_bt(cB, 40, wn * 32 + nf * 16 + ll, hb);
#pragma unroll
    for (int mf = 0; mf < 4; ++mf)
#pragma unroll
      for (int nf = 0; nf < 2; ++nf) acc[mf][nf] = wmma_bf16(af[mf], bfr[nf], acc[mf][nf]);

    wait_async0();   // next tile landed (overlapped with WMMA above)
    __syncthreads();
  }

#pragma unroll
  for (int mf = 0; mf < 4; ++mf)
#pragma unroll
    for (int nf = 0; nf < 2; ++nf)
#pragma unroll
      for (int r = 0; r < 8; ++r) {
        int m = m0 + wm * 64 + mf * 16 + r + hb * 8;  // C layout: M = r + half*8
        int e = n0 + wn * 32 + nf * 16 + ll;          // N = lane%16
        int which = e >> 10, rr = e & 1023, h = rr >> 6, i2 = rr & 63;
        int b = m >> 11, n = m & (NN - 1);
        size_t off = (((size_t)(b * HH + h)) * NN + n) * HD + i2;
        __bf16 v = (__bf16)acc[mf][nf][r];
        if (which == 0) Qb[off] = v;
        else if (which == 1) Kb[off] = v;
        else Vb[off] = v;
      }
}

// ---------------- fused flash attention ----------------
// grid: (B*H)*32 blocks, 128 threads (4 waves). Each block = 64 Q rows of one
// (b,h); each wave owns 16 Q rows. Stream K/V in 64-key blocks, online softmax.
// K/V tiles double-buffered via async global->LDS.
__global__ void __launch_bounds__(128) attn_kernel(
    const __bf16* __restrict__ Q, const __bf16* __restrict__ K,
    const __bf16* __restrict__ V, const float* __restrict__ bias,
    __bf16* __restrict__ ctx) {
  const int tid = threadIdx.x, lane = tid & 31, wave = tid >> 5;
  const int ll = lane & 15, hb = lane >> 4;
  const int bh = blockIdx.x >> 5;
  const int q0 = (blockIdx.x & 31) * 64;
  const int b = bh >> 4, h = bh & 15;

  const __bf16* Qp = Q + (size_t)bh * NN * HD;
  const __bf16* Kp = K + (size_t)bh * NN * HD;
  const __bf16* Vp = V + (size_t)bh * NN * HD;

  __shared__ __bf16 lQ[64 * 72];
  __shared__ __bf16 lK[2][64 * 72];
  __shared__ __bf16 lV[2][64 * 72];
  __shared__ __bf16 lP[4][16 * 72];  // per-wave C->A relayout scratch

  auto stage_kv = [&](int k0, int buf) {
    for (int vv = tid; vv < 512; vv += 128) {
      int row = vv >> 3, kv = (vv & 7) << 3;
      async_load_b128((unsigned)(size_t)&lK[buf][row * 72 + kv],
                      Kp + (size_t)(k0 + row) * HD + kv);
      async_load_b128((unsigned)(size_t)&lV[buf][row * 72 + kv],
                      Vp + (size_t)(k0 + row) * HD + kv);
    }
  };

  // stage Q + first K/V block
  for (int vv = tid; vv < 512; vv += 128) {
    int row = vv >> 3, kv = (vv & 7) << 3;
    async_load_b128((unsigned)(size_t)&lQ[row * 72 + kv],
                    Qp + (size_t)(q0 + row) * HD + kv);
  }
  stage_kv(0, 0);
  wait_async0();
  __syncthreads();

  v16bf aq[2];
#pragma unroll
  for (int c = 0; c < 2; ++c) aq[c] = frag_a(&lQ[c * 32], 72, wave * 16 + ll, hb);

  float mprev[8], lsum[8];
  v8f o[4];
#pragma unroll
  for (int r = 0; r < 8; ++r) { mprev[r] = -1e30f; lsum[r] = 0.f; }
#pragma unroll
  for (int nf = 0; nf < 4; ++nf) o[nf] = vzero();

  for (int kb = 0; kb < NN / 64; ++kb) {
    const int k0 = kb * 64;
    if (kb + 1 < NN / 64) stage_kv(k0 + 64, (kb + 1) & 1);
    const __bf16* cK = lK[kb & 1];
    const __bf16* cV = lV[kb & 1];

    // S = Q K^T (16x64 per wave)
    v8f s[4];
#pragma unroll
    for (int nf = 0; nf < 4; ++nf) {
      s[nf] = vzero();
#pragma unroll
      for (int c = 0; c < 2; ++c) {
        v16bf bk = frag_bt(&cK[c * 32], 72, nf * 16 + ll, hb);
        s[nf] = wmma_bf16(aq[c], bk, s[nf]);
      }
    }

    // scale + spatial bias; block row max
    float mblk[8];
#pragma unroll
    for (int r = 0; r < 8; ++r) mblk[r] = -1e30f;
#pragma unroll
    for (int nf = 0; nf < 4; ++nf) {
      const int kj = k0 + nf * 16 + ll;
#pragma unroll
      for (int r = 0; r < 8; ++r) {
        const int qi = q0 + wave * 16 + r + hb * 8;
        float val = s[nf][r] * SCALE + bias[(size_t)qi * NN + kj];
        s[nf][r] = val;
        mblk[r] = fmaxf(mblk[r], val);
      }
    }
#pragma unroll
    for (int off = 8; off >= 1; off >>= 1)
#pragma unroll
      for (int r = 0; r < 8; ++r) mblk[r] = fmaxf(mblk[r], __shfl_xor(mblk[r], off, 16));

    float corr[8], rsb[8];
#pragma unroll
    for (int r = 0; r < 8; ++r) {
      float mnew = fmaxf(mprev[r], mblk[r]);
      corr[r] = __expf(mprev[r] - mnew);
      mprev[r] = mnew;
      rsb[r] = 0.f;
    }

    // P = exp(S - m); stash bf16 P in per-wave LDS (C layout -> A layout)
#pragma unroll
    for (int nf = 0; nf < 4; ++nf)
#pragma unroll
      for (int r = 0; r < 8; ++r) {
        float p = __expf(s[nf][r] - mprev[r]);
        rsb[r] += p;
        lP[wave][(r + hb * 8) * 72 + nf * 16 + ll] = (__bf16)p;
      }
#pragma unroll
    for (int off = 8; off >= 1; off >>= 1)
#pragma unroll
      for (int r = 0; r < 8; ++r) rsb[r] += __shfl_xor(rsb[r], off, 16);
#pragma unroll
    for (int r = 0; r < 8; ++r) lsum[r] = lsum[r] * corr[r] + rsb[r];

    // rescale O, then O += P V
#pragma unroll
    for (int nf = 0; nf < 4; ++nf)
#pragma unroll
      for (int r = 0; r < 8; ++r) o[nf][r] *= corr[r];

    v16bf pa[2];
#pragma unroll
    for (int c = 0; c < 2; ++c) pa[c] = frag_a(&lP[wave][c * 32], 72, ll, hb);
#pragma unroll
    for (int nf = 0; nf < 4; ++nf)
#pragma unroll
      for (int c = 0; c < 2; ++c) {
        v16bf vb = frag_bn(&cV[(c * 32) * 72], 72, nf * 16 + ll, hb);
        o[nf] = wmma_bf16(pa[c], vb, o[nf]);
      }

    wait_async0();   // next K/V tiles landed (overlapped with compute above)
    __syncthreads();
  }

#pragma unroll
  for (int r = 0; r < 8; ++r) lsum[r] = 1.f / lsum[r];
#pragma unroll
  for (int nf = 0; nf < 4; ++nf)
#pragma unroll
    for (int r = 0; r < 8; ++r) {
      int n = q0 + wave * 16 + r + hb * 8;
      ctx[((size_t)(b * NN + n)) * DD + h * HD + nf * 16 + ll] =
          (__bf16)(o[nf][r] * lsum[r]);
    }
}

// ---------------- proj GEMM + bias + LayerNorm residual ----------------
__global__ void __launch_bounds__(256) proj_gemm_kernel(
    const __bf16* __restrict__ A, const __bf16* __restrict__ Bw,
    const float* __restrict__ x, const float* __restrict__ bproj,
    const float* __restrict__ gamma, const float* __restrict__ beta,
    const float* __restrict__ stats, float* __restrict__ out) {
  const int tid = threadIdx.x, lane = tid & 31, wave = tid >> 5;
  const int ll = lane & 15, hb = lane >> 4;
  const int wm = wave >> 2, wn = wave & 3;
  const int m0 = blockIdx.x * 128, n0 = blockIdx.y * 128;

  __shared__ __bf16 lA[2][128 * 40];
  __shared__ __bf16 lB[2][128 * 40];

  auto stage = [&](int kt, int buf) {
    for (int vv = tid; vv < 512; vv += 256) {
      int row = vv >> 2, kv = (vv & 3) << 3;
      async_load_b128((unsigned)(size_t)&lA[buf][row * 40 + kv],
                      A + (size_t)(m0 + row) * DD + kt + kv);
      async_load_b128((unsigned)(size_t)&lB[buf][row * 40 + kv],
                      Bw + (size_t)(n0 + row) * DD + kt + kv);
    }
  };

  v8f acc[4][2];
#pragma unroll
  for (int mf = 0; mf < 4; ++mf)
#pragma unroll
    for (int nf = 0; nf < 2; ++nf) acc[mf][nf] = vzero();

  stage(0, 0);
  wait_async0();
  __syncthreads();

  for (int t = 0; t < DD / 32; ++t) {
    if (t + 1 < DD / 32) stage((t + 1) * 32, (t + 1) & 1);
    const __bf16* cA = lA[t & 1];
    const __bf16* cB = lB[t & 1];

    v16bf af[4];
#pragma unroll
    for (int mf = 0; mf < 4; ++mf) af[mf] = frag_a(cA, 40, wm * 64 + mf * 16 + ll, hb);
    v16bf bfr[2];
#pragma unroll
    for (int nf = 0; nf < 2; ++nf) bfr[nf] = frag_bt(cB, 40, wn * 32 + nf * 16 + ll, hb);
#pragma unroll
    for (int mf = 0; mf < 4; ++mf)
#pragma unroll
      for (int nf = 0; nf < 2; ++nf) acc[mf][nf] = wmma_bf16(af[mf], bfr[nf], acc[mf][nf]);

    wait_async0();
    __syncthreads();
  }

#pragma unroll
  for (int mf = 0; mf < 4; ++mf)
#pragma unroll
    for (int nf = 0; nf < 2; ++nf)
#pragma unroll
      for (int r = 0; r < 8; ++r) {
        int m = m0 + wm * 64 + mf * 16 + r + hb * 8;
        int e = n0 + wn * 32 + nf * 16 + ll;
        float mu = stats[2 * m], rs = stats[2 * m + 1];
        float ln = (x[(size_t)m * DD + e] - mu) * rs * gamma[e] + beta[e];
        out[(size_t)m * DD + e] = acc[mf][nf][r] + bproj[e] + ln;
      }
}

// ---------------- workspace layout (bytes) ----------------
static const size_t OFF_XB    = 0;                          //  8 MB bf16 x
static const size_t OFF_WQKV  = 8ull << 20;                 //  6 MB bf16 Wqkv
static const size_t OFF_WPROJ = 14ull << 20;                //  2 MB bf16 Wproj
static const size_t OFF_Q     = 16ull << 20;                //  8 MB bf16 Q[b][h][n][hd]
static const size_t OFF_K     = 24ull << 20;                //  8 MB
static const size_t OFF_V     = 32ull << 20;                //  8 MB
static const size_t OFF_CTX   = 40ull << 20;                //  8 MB bf16 context
static const size_t OFF_BIAS  = 48ull << 20;                // 16 MB f32 bias (L2-resident)
static const size_t OFF_STATS = 64ull << 20;                // 32 KB f32 (mu,rstd)
static const size_t OFF_MAXD  = (64ull << 20) + (64ull << 10);

extern "C" void kernel_launch(void* const* d_in, const int* in_sizes, int n_in,
                              void* d_out, int out_size, void* d_ws, size_t ws_size,
                              hipStream_t stream) {
  (void)in_sizes; (void)n_in; (void)out_size; (void)ws_size;
  const float* x    = (const float*)d_in[0];
  const float* Wqkv = (const float*)d_in[1];
  const float* Wpj  = (const float*)d_in[2];
  const float* bpj  = (const float*)d_in[3];
  const float* gam  = (const float*)d_in[4];
  const float* bet  = (const float*)d_in[5];
  const float* ctr  = (const float*)d_in[6];
  float* out = (float*)d_out;

  char* ws = (char*)d_ws;
  __bf16* xb     = (__bf16*)(ws + OFF_XB);
  __bf16* wqkvb  = (__bf16*)(ws + OFF_WQKV);
  __bf16* wprojb = (__bf16*)(ws + OFF_WPROJ);
  __bf16* Qb     = (__bf16*)(ws + OFF_Q);
  __bf16* Kb     = (__bf16*)(ws + OFF_K);
  __bf16* Vb     = (__bf16*)(ws + OFF_V);
  __bf16* ctx    = (__bf16*)(ws + OFF_CTX);
  float*  bias   = (float*)(ws + OFF_BIAS);
  float*  stats  = (float*)(ws + OFF_STATS);
  unsigned* maxd = (unsigned*)(ws + OFF_MAXD);

  init_kernel<<<1, 1, 0, stream>>>(maxd);
  f32_to_bf16_kernel<<<(BB * NN * DD) / 256, 256, 0, stream>>>(x, xb, BB * NN * DD);
  f32_to_bf16_kernel<<<(3 * DD * DD) / 256, 256, 0, stream>>>(Wqkv, wqkvb, 3 * DD * DD);
  f32_to_bf16_kernel<<<(DD * DD) / 256, 256, 0, stream>>>(Wpj, wprojb, DD * DD);
  rowstats_kernel<<<BB * NN, 256, 0, stream>>>(x, stats);
  maxdist_kernel<<<(PP * PP) / 256, 256, 0, stream>>>(ctr, maxd);
  bias_kernel<<<(NN * NN) / 256, 256, 0, stream>>>(ctr, maxd, bias);
  qkv_gemm_kernel<<<dim3(32, 24), 256, 0, stream>>>(xb, wqkvb, Qb, Kb, Vb);
  attn_kernel<<<BB * HH * (NN / 64), 128, 0, stream>>>(Qb, Kb, Vb, bias, ctx);
  proj_gemm_kernel<<<dim3(32, 8), 256, 0, stream>>>(ctx, wprojb, x, bpj, gam, bet, stats, out);
}
